// LlamaQDecoderLayer_21947282883013
// MI455X (gfx1250) — matrix operand order repaired and checked by reading, not compiled, not run on hardware
//
#include <hip/hip_runtime.h>
#include <hip/hip_bf16.h>

// ---------------- constants ----------------
constexpr int T  = 2048;
constexpr int Hd = 4096;
constexpr int NH = 32;
constexpr int HDIM = 128;
constexpr int Iff = 11008;
constexpr int GRP = 128;

// ---------------- WMMA / TDM types ----------------
typedef __attribute__((ext_vector_type(16))) __bf16 v16bf;
typedef __attribute__((ext_vector_type(8)))  float  v8f;
typedef __attribute__((ext_vector_type(4)))  unsigned int u32x4;
typedef __attribute__((ext_vector_type(8)))  unsigned int u32x8;

union AFrag { unsigned int u[8]; v16bf v; };
union Acc   { float f[8]; v8f v; };

// float -> bf16 (round to nearest even), returned as raw u16
static __device__ __forceinline__ unsigned short f2bf(float f) {
  unsigned int u = __float_as_uint(f);
  u = (u + 0x7FFFu + ((u >> 16) & 1u)) >> 16;
  return (unsigned short)u;
}

// ================= RMSNorm (fp32 in -> bf16 out) =================
__global__ __launch_bounds__(256) void rmsnorm_kernel(
    const float* __restrict__ x, const float* __restrict__ w,
    unsigned short* __restrict__ out) {
  int row = blockIdx.x;
  const float* xr = x + (size_t)row * Hd;
  float ss = 0.f;
  for (int i = threadIdx.x; i < Hd; i += 256) { float v = xr[i]; ss += v * v; }
  #pragma unroll
  for (int m = 16; m > 0; m >>= 1) ss += __shfl_xor(ss, m, 32);
  __shared__ float red[8];
  if ((threadIdx.x & 31) == 0) red[threadIdx.x >> 5] = ss;
  __syncthreads();
  if (threadIdx.x < 8) {
    float v = red[threadIdx.x];
    #pragma unroll
    for (int m = 4; m > 0; m >>= 1) v += __shfl_xor(v, m, 32);
    if (threadIdx.x == 0) red[0] = v;
  }
  __syncthreads();
  float inv = rsqrtf(red[0] / (float)Hd + 1e-6f);
  unsigned short* outr = out + (size_t)row * Hd;
  for (int i = threadIdx.x; i < Hd; i += 256) outr[i] = f2bf(xr[i] * inv * w[i]);
}

// ================= fused AWQ-dequant GEMM (bf16 WMMA, TDM A-staging) ======
// C[M,N] = Abf16[M,K] @ dequant(qw,scales,qz)[K,N]
// block tile: 64(M) x 256(N) x 32(K); 8 waves = 2x4; wave tile 32x64
// (2 A frags + 4 B frags -> 8 WMMA per K-step per wave)
// A tile staged by the Tensor Data Mover (double buffered, wave0-issued).
__global__ __launch_bounds__(256) void gemm_awq_wmma(
    const unsigned short* __restrict__ Abf, const int* __restrict__ qw,
    const float* __restrict__ scales, const int* __restrict__ qz,
    float* __restrict__ C, int M, int N, int K) {
  __shared__ unsigned short As[2][64][32];   // 8KB, TDM double buffer
  __shared__ unsigned short Bs[256][32];     // 16KB, weights, K-contig per column

  const int m0 = blockIdx.y * 64;
  const int n0 = blockIdx.x * 256;
  const int tid  = threadIdx.x;
  const int wave = tid >> 5, lane = tid & 31;
  const int wm = wave >> 2, wn = wave & 3;   // wave tile: rows wm*32, cols wn*64
  const int halfw = lane >> 4;
  const int l16   = lane & 15;

  // ---- TDM descriptor group1 (loop invariant): 2D bf16 tensor [M][K], tile 32x64
  u32x8 g1;
  g1[0] = 1u << 16;                                             // data_size = 2 bytes
  g1[1] = ((unsigned)K & 0xffffu) << 16;                        // tensor_dim0 lo
  g1[2] = (((unsigned)K >> 16) & 0xffffu) |
          (((unsigned)M & 0xffffu) << 16);                      // dim0 hi | dim1 lo
  g1[3] = (((unsigned)M >> 16) & 0xffffu) | (32u << 16);        // dim1 hi | tile_dim0=32
  g1[4] = 64u;                                                  // tile_dim1=64
  g1[5] = (unsigned)K;                                          // dim0_stride lo32
  g1[6] = 0u;                                                   // dim0_stride hi16
  g1[7] = 0u;
  const unsigned long long gbase =
      (unsigned long long)(size_t)(Abf + (size_t)m0 * K);       // row m0, col 0 (bytes)

  Acc acc[2][4];
  #pragma unroll
  for (int a = 0; a < 2; ++a)
    #pragma unroll
    for (int b = 0; b < 4; ++b)
      #pragma unroll
      for (int j = 0; j < 8; ++j) acc[a][b].f[j] = 0.f;

  // ---- prime: issue DMA of first A tile into As[0]
  if (wave == 0) {
    unsigned long long ga = gbase;                  // k0 = 0
    u32x4 g0;
    g0[0] = 1u;                                     // count=1
    g0[1] = (unsigned)(size_t)(&As[0][0][0]);       // lds_addr (bytes)
    g0[2] = (unsigned)ga;                           // global_addr lo
    g0[3] = (unsigned)(ga >> 32) | (2u << 30);      // global_addr hi | type=2
    asm volatile("tensor_load_to_lds %0, %1" :: "s"(g0), "s"(g1) : "memory");
  }

  // B staging: thread owns one output column, walks K with 32-bit offsets
  const int ncol = tid;                              // 0..255
  unsigned boff0 = (unsigned)(n0 + ncol);            // element offset at k=0

  int buf = 0;
  for (int k0 = 0; k0 < K; k0 += 32) {
    __syncthreads();   // previous compute finished reading As/Bs

    // ---- stage B tile: dequant int4 -> bf16, column-major-in-K, packed b128 ----
    {
      unsigned gi = (unsigned)(k0 / GRP) * (unsigned)N + boff0;
      float sc = scales[gi];
      float zp = (float)qz[gi];
      unsigned off = (unsigned)k0 * (unsigned)N + boff0;
      #pragma unroll
      for (int q = 0; q < 4; ++q) {                  // 4 x (8 bf16 = 16B) stores
        u32x4 pk;
        #pragma unroll
        for (int e = 0; e < 4; ++e) {
          float w0 = ((float)qw[off] - zp) * sc; off += (unsigned)N;
          float w1 = ((float)qw[off] - zp) * sc; off += (unsigned)N;
          pk[e] = (unsigned)f2bf(w0) | ((unsigned)f2bf(w1) << 16);
        }
        *(u32x4*)(&Bs[ncol][q * 8]) = pk;
      }
      if (k0 + 32 < K)                               // prefetch next K-tile of weights
        __builtin_prefetch(qw + off, 0, 0);          // off == (k0+32)*N + n0 + ncol
    }

    if (wave == 0) __builtin_amdgcn_s_wait_tensorcnt(0);  // As[buf] landed
    __syncthreads();                                      // publish As[buf] + Bs

    // ---- issue DMA for next A tile into As[buf^1] (overlaps compute) ----
    if (wave == 0 && (k0 + 32) < K) {
      unsigned long long ga = gbase + (unsigned)(k0 + 32) * 2ull;
      u32x4 g0;
      g0[0] = 1u;
      g0[1] = (unsigned)(size_t)(&As[buf ^ 1][0][0]);
      g0[2] = (unsigned)ga;
      g0[3] = (unsigned)(ga >> 32) | (2u << 30);
      asm volatile("tensor_load_to_lds %0, %1" :: "s"(g0), "s"(g1) : "memory");
    }

    // ---- fragments + WMMA (2 x 4 = 8 WMMA per K-step) ----
    AFrag a[2];
    #pragma unroll
    for (int mi = 0; mi < 2; ++mi) {
      const unsigned short* ar = &As[buf][wm * 32 + mi * 16 + l16][0];
      #pragma unroll
      for (int i = 0; i < 8; ++i) {
        int k = ((i < 4) ? 2 * i : 16 + 2 * (i - 4)) + halfw * 8;
        a[mi].u[i] = *(const unsigned int*)(ar + k);
      }
    }
    #pragma unroll
    for (int ni = 0; ni < 4; ++ni) {
      const unsigned short* br = &Bs[wn * 64 + ni * 16 + l16][0];
      AFrag bf;
      #pragma unroll
      for (int j = 0; j < 8; ++j)
        bf.u[j] = *(const unsigned int*)(br + ((j < 4) ? 2 * j : 16 + 2 * (j - 4)) + halfw * 8);
      #pragma unroll
      for (int mi = 0; mi < 2; ++mi)
        acc[mi][ni].v = __builtin_amdgcn_wmma_f32_16x16x32_bf16(
            false, a[mi].v, false, bf.v, (short)0, acc[mi][ni].v, false, false);
    }
    buf ^= 1;
  }

  // ---- store C ----
  #pragma unroll
  for (int mi = 0; mi < 2; ++mi)
    #pragma unroll
    for (int ni = 0; ni < 4; ++ni) {
      int mb = m0 + wm * 32 + mi * 16 + halfw * 8;
      int nb = n0 + wn * 64 + ni * 16 + l16;
      #pragma unroll
      for (int j = 0; j < 8; ++j)
        C[(size_t)(mb + j) * N + nb] = acc[mi][ni].f[j];
    }
}

// ================= RoPE (in-place on qkv fp32, q and k halves) ============
__global__ __launch_bounds__(256) void rope_kernel(
    float* __restrict__ qkv, const long long* __restrict__ pos) {
  int idx = blockIdx.x * 256 + threadIdx.x;   // T*NH*64 threads
  int d = idx & 63;
  int h = (idx >> 6) & 31;
  int t = idx >> 11;
  if (t >= T) return;
  float p = (float)pos[t];
  float ang = p * __expf(-(float)(2 * d) * (9.210340371976184f / 128.0f));
  float s, c;
  __sincosf(ang, &s, &c);
  #pragma unroll
  for (int qk = 0; qk < 2; ++qk) {
    float* base = qkv + (size_t)t * (3 * Hd) + qk * Hd + h * HDIM;
    float x1 = base[d], x2 = base[d + 64];
    base[d]      = x1 * c - x2 * s;
    base[d + 64] = x2 * c + x1 * s;
  }
}

// ================= flash attention (one wave per head x 16-row Q tile) ====
__global__ __launch_bounds__(32) void attn_wmma_kernel(
    const float* __restrict__ qkv, unsigned short* __restrict__ out) {
  __shared__ unsigned short Qs[16][HDIM];   // 4KB
  __shared__ unsigned short Ks[32][HDIM];   // 8KB  [key][hd]
  __shared__ unsigned short Vs[HDIM][32];   // 8KB  [hd][key] (transposed)
  __shared__ unsigned short Ps[16][32];     // 1KB  P staging (row-major)

  const int h  = blockIdx.x;
  const int m0 = blockIdx.y * 16;
  const int lane = threadIdx.x;
  const int l16 = lane & 15, halfw = lane >> 4;
  const float scale = 0.08838834764831845f;   // HD^-0.5

  for (int i = lane; i < 16 * HDIM; i += 32) {
    int r = i >> 7, d = i & 127;
    Qs[r][d] = f2bf(qkv[(size_t)(m0 + r) * (3 * Hd) + h * HDIM + d] * scale);
  }
  __syncthreads();

  AFrag qf[4];
  #pragma unroll
  for (int kc = 0; kc < 4; ++kc)
    #pragma unroll
    for (int i = 0; i < 8; ++i) {
      int k = kc * 32 + ((i < 4) ? 2 * i : 16 + 2 * (i - 4)) + halfw * 8;
      qf[kc].u[i] = *(const unsigned int*)(&Qs[l16][k]);
    }

  Acc o[8];
  float mstate[8], lstate[8];
  #pragma unroll
  for (int j = 0; j < 8; ++j) {
    mstate[j] = -3.0e38f; lstate[j] = 0.f;
    #pragma unroll
    for (int nc = 0; nc < 8; ++nc) o[nc].f[j] = 0.f;
  }

  for (int j0 = 0; j0 <= m0 + 15; j0 += 32) {
    for (int i = lane; i < 32 * HDIM; i += 32) {
      int r = i >> 7, d = i & 127;
      size_t rowb = (size_t)(j0 + r) * (3 * Hd) + h * HDIM + d;
      Ks[r][d] = f2bf(qkv[rowb + Hd]);       // K
      Vs[d][r] = f2bf(qkv[rowb + 2 * Hd]);   // V transposed
    }
    __syncthreads();

    // S = Q @ K^T  (two 16x16 frags over 32 keys)
    Acc s[2];
    #pragma unroll
    for (int kn = 0; kn < 2; ++kn) {
      #pragma unroll
      for (int j = 0; j < 8; ++j) s[kn].f[j] = 0.f;
      const unsigned short* kr = &Ks[kn * 16 + l16][0];
      #pragma unroll
      for (int kc = 0; kc < 4; ++kc) {
        AFrag bf;
        #pragma unroll
        for (int j = 0; j < 8; ++j)
          bf.u[j] = *(const unsigned int*)(kr + kc * 32 + ((j < 4) ? 2 * j : 16 + 2 * (j - 4)) + halfw * 8);
        s[kn].v = __builtin_amdgcn_wmma_f32_16x16x32_bf16(
            false, qf[kc].v, false, bf.v, (short)0, s[kn].v, false, false);
      }
    }

    // causal mask + online softmax (row j+8*half lives in one 16-lane half)
    #pragma unroll
    for (int j = 0; j < 8; ++j) {
      int qrow = m0 + j + 8 * halfw;
      float s0 = (j0 + l16      <= qrow) ? s[0].f[j] : -3.0e38f;
      float s1 = (j0 + 16 + l16 <= qrow) ? s[1].f[j] : -3.0e38f;
      float rm = fmaxf(s0, s1);
      #pragma unroll
      for (int msk = 1; msk < 16; msk <<= 1) rm = fmaxf(rm, __shfl_xor(rm, msk, 32));
      float mnew = fmaxf(mstate[j], rm);
      float p0 = __expf(s0 - mnew), p1 = __expf(s1 - mnew);
      float rs = p0 + p1;
      #pragma unroll
      for (int msk = 1; msk < 16; msk <<= 1) rs += __shfl_xor(rs, msk, 32);
      float alpha = __expf(mstate[j] - mnew);
      lstate[j] = lstate[j] * alpha + rs;
      mstate[j] = mnew;
      #pragma unroll
      for (int nc = 0; nc < 8; ++nc) o[nc].f[j] *= alpha;
      Ps[j + 8 * halfw][l16]      = f2bf(p0);
      Ps[j + 8 * halfw][16 + l16] = f2bf(p1);
    }
    __syncthreads();

    // P fragment (A layout) then O += P @ V
    AFrag pf;
    #pragma unroll
    for (int i = 0; i < 8; ++i) {
      int k = ((i < 4) ? 2 * i : 16 + 2 * (i - 4)) + halfw * 8;
      pf.u[i] = *(const unsigned int*)(&Ps[l16][k]);
    }
    #pragma unroll
    for (int nc = 0; nc < 8; ++nc) {
      AFrag bf;
      const unsigned short* vr = &Vs[nc * 16 + l16][0];
      #pragma unroll
      for (int j = 0; j < 8; ++j)
        bf.u[j] = *(const unsigned int*)(vr + ((j < 4) ? 2 * j : 16 + 2 * (j - 4)) + halfw * 8);
      o[nc].v = __builtin_amdgcn_wmma_f32_16x16x32_bf16(
          false, pf.v, false, bf.v, (short)0, o[nc].v, false, false);
    }
    __syncthreads();
  }

  // epilogue: divide by l, write bf16 [T, H]
  #pragma unroll
  for (int j = 0; j < 8; ++j) {
    float inv = 1.0f / lstate[j];
    int row = m0 + j + 8 * halfw;
    #pragma unroll
    for (int nc = 0; nc < 8; ++nc)
      out[(size_t)row * Hd + h * HDIM + nc * 16 + l16] = f2bf(o[nc].f[j] * inv);
  }
}

// ================= elementwise =================
__global__ __launch_bounds__(256) void add_kernel(
    const float* __restrict__ a, const float* __restrict__ b,
    float* __restrict__ o, int n) {
  int i = blockIdx.x * 256 + threadIdx.x;
  if (i < n) o[i] = a[i] + b[i];
}

__global__ __launch_bounds__(256) void silu_mul_kernel(
    const float* __restrict__ gu, unsigned short* __restrict__ o) {
  int i = blockIdx.x * 256 + threadIdx.x;
  if (i >= T * Iff) return;
  int t = i / Iff, c = i - t * Iff;
  const float* row = gu + (size_t)t * (2 * Iff);
  float g = row[c], u = row[Iff + c];
  o[i] = f2bf(g / (1.f + __expf(-g)) * u);
}

// ================= launch =================
extern "C" void kernel_launch(void* const* d_in, const int* in_sizes, int n_in,
                              void* d_out, int out_size, void* d_ws, size_t ws_size,
                              hipStream_t stream) {
  const long long* positions = (const long long*)d_in[0];
  const float* hidden   = (const float*)d_in[1];
  const float* in_ln_w  = (const float*)d_in[2];
  const float* post_ln_w= (const float*)d_in[3];
  const int*   qkv_qw   = (const int*)d_in[4];
  const float* qkv_sc   = (const float*)d_in[5];
  const int*   qkv_qz   = (const int*)d_in[6];
  const int*   o_qw     = (const int*)d_in[7];
  const float* o_sc     = (const float*)d_in[8];
  const int*   o_qz     = (const int*)d_in[9];
  const int*   gu_qw    = (const int*)d_in[10];
  const float* gu_sc    = (const float*)d_in[11];
  const int*   gu_qz    = (const int*)d_in[12];
  const int*   dn_qw    = (const int*)d_in[13];
  const float* dn_sc    = (const float*)d_in[14];
  const int*   dn_qz    = (const int*)d_in[15];
  float* out = (float*)d_out;

  char* p = (char*)d_ws;
  unsigned short* xnorm = (unsigned short*)p; p += (size_t)T * Hd * 2;       // bf16
  float*          qkv   = (float*)p;          p += (size_t)T * 3 * Hd * 4;
  unsigned short* attnb = (unsigned short*)p; p += (size_t)T * Hd * 2;       // bf16
  float*          o_out = (float*)p;          p += (size_t)T * Hd * 4;
  float*          hid2  = (float*)p;          p += (size_t)T * Hd * 4;
  float*          gateup= (float*)p;          p += (size_t)T * 2 * Iff * 4;
  unsigned short* act   = (unsigned short*)p; p += (size_t)T * Iff * 2;      // bf16
  float*          dnout = o_out;   // o_out is dead after the first residual add

  // 1) RMSNorm -> bf16
  rmsnorm_kernel<<<T, 256, 0, stream>>>(hidden, in_ln_w, xnorm);
  // 2) QKV projection (M=2048, N=12288, K=4096)
  gemm_awq_wmma<<<dim3(3 * Hd / 256, T / 64), 256, 0, stream>>>(
      xnorm, qkv_qw, qkv_sc, qkv_qz, qkv, T, 3 * Hd, Hd);
  // 3) RoPE on q,k
  rope_kernel<<<(T * NH * 64) / 256, 256, 0, stream>>>(qkv, positions);
  // 4) causal flash attention -> bf16
  attn_wmma_kernel<<<dim3(NH, T / 16), 32, 0, stream>>>(qkv, attnb);
  // 5) O projection (N=4096, K=4096)
  gemm_awq_wmma<<<dim3(Hd / 256, T / 64), 256, 0, stream>>>(
      attnb, o_qw, o_sc, o_qz, o_out, T, Hd, Hd);
  // 6) residual
  add_kernel<<<(T * Hd) / 256, 256, 0, stream>>>(hidden, o_out, hid2, T * Hd);
  // 7) RMSNorm 2 -> bf16
  rmsnorm_kernel<<<T, 256, 0, stream>>>(hid2, post_ln_w, xnorm);
  // 8) gate/up projection (N=22016, K=4096)
  gemm_awq_wmma<<<dim3(2 * Iff / 256, T / 64), 256, 0, stream>>>(
      xnorm, gu_qw, gu_sc, gu_qz, gateup, T, 2 * Iff, Hd);
  // 9) SiLU(gate) * up -> bf16
  silu_mul_kernel<<<(T * Iff + 255) / 256, 256, 0, stream>>>(gateup, act);
  // 10) down projection (N=4096, K=11008)
  gemm_awq_wmma<<<dim3(Hd / 256, T / 64), 256, 0, stream>>>(
      act, dn_qw, dn_sc, dn_qz, dnout, T, Hd, Iff);
  // 11) final residual -> output
  add_kernel<<<(T * Hd) / 256, 256, 0, stream>>>(hid2, dnout, out, T * Hd);
}
